// PCACanonizationNetwork_28716151341736
// MI455X (gfx1250) — matrix operand order repaired and checked by reading, not compiled, not run on hardware
//
#include <hip/hip_runtime.h>
#include <hip/hip_bf16.h>
#include <math.h>

typedef __attribute__((ext_vector_type(16))) _Float16 v16h;
typedef __attribute__((ext_vector_type(8)))  _Float16 v8h;
typedef __attribute__((ext_vector_type(8)))  float    v8f;
typedef __attribute__((ext_vector_type(4)))  unsigned int u32x4;
typedef __attribute__((ext_vector_type(8)))  unsigned int u32x8;

#define B_IMG   2048
#define HW      784          // 28*28
#define C1      32
#define C2      64
#define K2      288          // 32*9
#define P2      196          // 14*14
#define M2      (B_IMG*P2)   // 401408
#define KF      12544        // 64*196
#define NPAD    16

// ---------------- workspace layout (bytes) ----------------
#define OFF_THETA 0
#define OFF_H1    8192
#define OFF_OUT2  (OFF_H1  + 102760448)
#define OFF_W2T   (OFF_OUT2 + 51380224)
#define OFF_PWT   (OFF_W2T + 36864)

// ================= prep: cast conv2 weights to f16, K permuted =================
// K order: k = ky*96 + kx*32 + c   (chunk kc = ky*3+kx holds channels 0..31)
__global__ void k_prep_w2t(const float* __restrict__ w2, _Float16* __restrict__ w2t) {
    int i = blockIdx.x * 256 + threadIdx.x;
    if (i < C2 * K2) {
        int n = i / K2, k = i % K2;          // src: n*288 + c*9 + ky*3 + kx
        int c = k / 9, rr = k % 9;
        int ky = rr / 3, kx = rr % 3;
        w2t[n * K2 + ky * 96 + kx * 32 + c] = (_Float16)w2[i];
    }
}

// pred_w transposed+padded with K permuted to kk = p*64 + c  (matches out2 layout)
__global__ void k_prep_pwt(const float* __restrict__ pw, _Float16* __restrict__ pwt) {
    int i = blockIdx.x * 256 + threadIdx.x;
    if (i < NPAD * KF) {
        int n = i / KF, kk = i % KF;
        int p = kk >> 6, c = kk & 63;        // kk = p*64 + c
        pwt[i] = (n < 10) ? (_Float16)pw[(c * P2 + p) * 10 + n] : (_Float16)0.0f;
    }
}

// ================= angles: masked 2x2 PCA per image =================
__global__ void k_angles(const float* __restrict__ x, float* __restrict__ theta) {
    const int b = blockIdx.x;
    const float* img = x + (size_t)b * HW;
    float s1 = 0.f, sx = 0.f, sy = 0.f, sxx = 0.f, sxy = 0.f, syy = 0.f;
    for (int p = threadIdx.x; p < HW; p += blockDim.x) {
        int i = p / 28, j = p % 28;
        float m  = img[p] > 0.5f ? 1.0f : 0.0f;
        float gx = -14.0f + (28.0f / 27.0f) * (float)j;
        float gy =  14.0f - (28.0f / 27.0f) * (float)i;
        s1 += m; sx += m * gx; sy += m * gy;
        sxx += m * gx * gx; sxy += m * gx * gy; syy += m * gy * gy;
    }
    #pragma unroll
    for (int o = 16; o > 0; o >>= 1) {
        s1  += __shfl_xor(s1,  o);  sx  += __shfl_xor(sx,  o);
        sy  += __shfl_xor(sy,  o);  sxx += __shfl_xor(sxx, o);
        sxy += __shfl_xor(sxy, o);  syy += __shfl_xor(syy, o);
    }
    __shared__ float red[8][6];
    int wid = threadIdx.x >> 5, lane = threadIdx.x & 31;
    if (lane == 0) {
        red[wid][0] = s1;  red[wid][1] = sx;  red[wid][2] = sy;
        red[wid][3] = sxx; red[wid][4] = sxy; red[wid][5] = syy;
    }
    __syncthreads();
    if (threadIdx.x == 0) {
        float S1 = 0, Sx = 0, Sy = 0, Sxx = 0, Sxy = 0, Syy = 0;
        for (int w = 0; w < 8; ++w) {
            S1 += red[w][0]; Sx += red[w][1]; Sy += red[w][2];
            Sxx += red[w][3]; Sxy += red[w][4]; Syy += red[w][5];
        }
        float n  = fmaxf(S1, 1.0f);
        float mx = Sx / n, my = Sy / n;
        float a  = Sxx - 2.f * mx * Sx + mx * mx * S1;
        float bb = Sxy - mx * Sy - my * Sx + mx * my * S1;
        float c  = Syy - 2.f * my * Sy + my * my * S1;
        float lam = 0.5f * (a + c) + sqrtf(0.25f * (a - c) * (a - c) + bb * bb);
        float v1x = bb, v1y = lam - a, v2x = lam - c, v2y = bb;
        float q1 = v1x * v1x + v1y * v1y, q2 = v2x * v2x + v2y * v2y;
        float vx = (q1 >= q2) ? v1x : v2x;
        float vy = (q1 >= q2) ? v1y : v2y;
        if (vx * vx + vy * vy < 1e-12f) { vx = 1.0f; vy = 0.0f; }
        theta[b] = -atan2f(vy, vx);   // rotate by -angle
    }
}

// ================= rotate (bilinear) + conv1 + relu, fused per image =================
__global__ void k_rot_conv1(const float* __restrict__ x, const float* __restrict__ theta,
                            const float* __restrict__ w1, const float* __restrict__ b1,
                            _Float16* __restrict__ h1) {
    __shared__ float rimg[HW];
    __shared__ float sw[C1 * 9];
    __shared__ float sb[C1];
    const int b = blockIdx.x;
    const float th = theta[b];
    const float cs = cosf(th), sn = sinf(th);
    const float* img = x + (size_t)b * HW;
    if (threadIdx.x < C1 * 9) sw[threadIdx.x] = w1[threadIdx.x];
    if (threadIdx.x < C1)     sb[threadIdx.x] = b1[threadIdx.x];

    for (int p = threadIdx.x; p < HW; p += blockDim.x) {
        int yy = p / 28, xx = p % 28;
        float dxp = (float)xx - 13.5f, dyp = (float)yy - 13.5f;
        float sxf =  cs * dxp + sn * dyp + 13.5f;
        float syf = -sn * dxp + cs * dyp + 13.5f;
        float x0 = floorf(sxf), y0 = floorf(syf);
        float wx1 = sxf - x0, wx0 = 1.f - wx1;
        float wy1 = syf - y0, wy0 = 1.f - wy1;
        int ix0 = (int)x0, iy0 = (int)y0;
        int yc0 = min(max(iy0, 0), 27),     yc1 = min(max(iy0 + 1, 0), 27);
        int xc0 = min(max(ix0, 0), 27),     xc1 = min(max(ix0 + 1, 0), 27);
        bool vy0 = (unsigned)iy0 < 28u,     vy1 = (unsigned)(iy0 + 1) < 28u;
        bool vx0 = (unsigned)ix0 < 28u,     vx1 = (unsigned)(ix0 + 1) < 28u;
        float v00 = (vy0 && vx0) ? img[yc0 * 28 + xc0] : 0.f;
        float v01 = (vy0 && vx1) ? img[yc0 * 28 + xc1] : 0.f;
        float v10 = (vy1 && vx0) ? img[yc1 * 28 + xc0] : 0.f;
        float v11 = (vy1 && vx1) ? img[yc1 * 28 + xc1] : 0.f;
        rimg[p] = v00 * wy0 * wx0 + v01 * wy0 * wx1 + v10 * wy1 * wx0 + v11 * wy1 * wx1;
    }
    __syncthreads();

    _Float16* o = h1 + (size_t)b * (C1 * HW);
    for (int t = threadIdx.x; t < C1 * HW; t += blockDim.x) {
        int c = t / HW, p = t % HW, y = p / 28, xc = p % 28;
        float acc = sb[c];
        #pragma unroll
        for (int ky = 0; ky < 3; ++ky) {
            int iy = y + ky - 1;
            if ((unsigned)iy < 28u) {
                #pragma unroll
                for (int kx = 0; kx < 3; ++kx) {
                    int ix = xc + kx - 1;
                    if ((unsigned)ix < 28u)
                        acc += sw[c * 9 + ky * 3 + kx] * rimg[iy * 28 + ix];
                }
            }
        }
        o[t] = (_Float16)fmaxf(acc, 0.0f);
    }
}

// ================= conv2 as implicit GEMM via WMMA f16 =================
// M = 401408 (b*196+p), K = 288 permuted as ky*96+kx*32+c, N = 64
// Weights (36KB) staged in LDS once per block via Tensor Data Mover.
// A gather: one base pointer per lane, 144 immediate-offset loads.
// Output written as out2[b][p*64+c]  (coalesced across lanes).
__global__ void __launch_bounds__(256) k_conv2_wmma(
        const _Float16* __restrict__ h1, const _Float16* __restrict__ w2t,
        const float* __restrict__ c2b, _Float16* __restrict__ out2) {
    __shared__ _Float16 sB[C2 * K2];                 // 36,864 B

    // ---- TDM: DMA w2t (contiguous 9216 dwords) into LDS, wave 0 issues ----
    if (threadIdx.x < 32) {
        const unsigned DW = (C2 * K2 * 2) / 4;       // 9216 dwords
        unsigned long long ga = (unsigned long long)(size_t)w2t;
        unsigned int ldsoff = (unsigned int)(unsigned long long)(size_t)&sB[0];
        u32x4 g0;
        g0[0] = 1u;                                   // count=1, no gather
        g0[1] = ldsoff;                               // lds_addr
        g0[2] = (unsigned int)(ga & 0xffffffffull);   // global_addr[31:0]
        g0[3] = (unsigned int)((ga >> 32) & 0x01ffffffull) | (2u << 30); // addr[56:32] | type=2
        u32x8 g1;
        g1[0] = 2u << 16;                             // data_size=4B
        g1[1] = (DW & 0xffffu) << 16;                 // tensor_dim0[15:0]
        g1[2] = (DW >> 16) | (1u << 16);              // tensor_dim0[31:16] | tensor_dim1=1
        g1[3] = (DW & 0xffffu) << 16;                 // tile_dim0 = 9216
        g1[4] = 1u;                                   // tile_dim1=1, tile_dim2=0
        g1[5] = DW;                                   // tensor_dim0_stride[31:0]
        g1[6] = (DW & 0xffffu) << 16;                 // stride0 hi | tensor_dim1_stride[15:0]
        g1[7] = (DW >> 16);                           // tensor_dim1_stride[47:16]
        asm volatile("tensor_load_to_lds %0, %1" :: "s"(g0), "s"(g1) : "memory");
        __builtin_amdgcn_s_wait_tensorcnt(0);
    }
    __syncthreads();

    const int wid  = threadIdx.x >> 5;
    const int lane = threadIdx.x & 31;
    const int tile = blockIdx.x * 8 + wid;           // exact: 25088 tiles
    const int mbase = tile * 16;
    const int r    = lane & 15;
    const int b0   = mbase / P2;
    const int p0   = mbase - b0 * P2;
    // A-row (per lane): 16 rows cross at most one image boundary
    int pr = p0 + r, br = b0;
    if (pr >= P2) { br += 1; pr -= P2; }
    const int oy   = pr / 14, ox = pr - (pr / 14) * 14;
    const int abase = (lane & 16) ? 8  : 0;          // A-fragment channel sub-base
    const int kg    = (lane & 16) ? 8  : 0;          // B-fragment K group
    // single base pointer: channel abase, spatial (2oy-1, 2ox-1)
    const _Float16* hb = h1 + (size_t)br * (C1 * HW)
                        + (size_t)abase * HW + (2 * oy - 1) * 28 + (2 * ox - 1);
    const bool oyok = (oy > 0);                      // iy=-1 only possible at ky=0
    const bool oxok = (ox > 0);                      // ix=-1 only possible at kx=0

    v8f acc[4] = {};
    #pragma unroll
    for (int ky = 0; ky < 3; ++ky) {
        #pragma unroll
        for (int kx = 0; kx < 3; ++kx) {
            const int kc = ky * 3 + kx;
            const bool ok = ((ky > 0) || oyok) && ((kx > 0) || oxok); // folds for 4 chunks
            // --- A fragment: 16 immediate-offset loads off one base ---
            v16h af;
            #pragma unroll
            for (int e = 0; e < 16; ++e) {
                const int ce = (e < 8) ? e : (8 + e);            // channel within half
                _Float16 v = hb[ce * HW + ky * 28 + kx];
                af[e] = ok ? v : (_Float16)0.0f;
            }
            // --- all four B fragments (two 8-half runs per lane, LDS) ---
            v16h bf[4];
            #pragma unroll
            for (int nt = 0; nt < 4; ++nt) {
                const _Float16* wp = sB + (nt * 16 + r) * K2 + kc * 32 + kg;
                v8h blo = *(const v8h*)(wp);
                v8h bhi = *(const v8h*)(wp + 16);
                #pragma unroll
                for (int e = 0; e < 8; ++e) { bf[nt][e] = blo[e]; bf[nt][8 + e] = bhi[e]; }
            }
            #pragma unroll
            for (int nt = 0; nt < 4; ++nt) {
                acc[nt] = __builtin_amdgcn_wmma_f32_16x16x32_f16(
                    false, af, false, bf[nt], (short)0, acc[nt], false, false);
            }
        }
    }
    // --- bias + relu + coalesced store: out2[b][p*64 + c] ---
    float bias[4];
    #pragma unroll
    for (int nt = 0; nt < 4; ++nt) bias[nt] = c2b[nt * 16 + r];
    #pragma unroll
    for (int i = 0; i < 8; ++i) {
        int off = i + ((lane >> 4) << 3);
        int pp = p0 + off, bb = b0;
        if (pp >= P2) { bb += 1; pp -= P2; }         // at most one wrap in 16 rows
        size_t base = (size_t)bb * KF + (size_t)pp * 64;
        #pragma unroll
        for (int nt = 0; nt < 4; ++nt) {
            float v = fmaxf(acc[nt][i] + bias[nt], 0.0f);
            out2[base + nt * 16 + r] = (_Float16)v;
        }
    }
}

// ================= final linear via WMMA f16: [2048,12544] x [12544,10] =================
// K dimension uses the permuted order kk = p*64+c on BOTH operands.
__global__ void __launch_bounds__(256) k_final_wmma(
        const _Float16* __restrict__ out2, const _Float16* __restrict__ pwt,
        const float* __restrict__ predb, float* __restrict__ out) {
    const int wid  = threadIdx.x >> 5;
    const int lane = threadIdx.x & 31;
    const int tile = blockIdx.x * 8 + wid;           // 128 tiles exact
    const int mbase = tile * 16;
    const int r    = lane & 15;
    const int abase = (lane & 16) ? 8  : 0;
    const int kg    = (lane & 16) ? 8  : 0;
    const _Float16* arow = out2 + (size_t)(mbase + r) * KF;
    const _Float16* brow = pwt  + (size_t)r * KF;

    v8f acc = {};
    #pragma unroll 4
    for (int kc = 0; kc < KF / 32; ++kc) {
        const _Float16* ap = arow + kc * 32 + abase;
        v8h alo = *(const v8h*)(ap);
        v8h ahi = *(const v8h*)(ap + 16);
        v16h af;
        #pragma unroll
        for (int e = 0; e < 8; ++e) { af[e] = alo[e]; af[8 + e] = ahi[e]; }
        const _Float16* bp = brow + kc * 32 + kg;
        v8h blo = *(const v8h*)(bp);
        v8h bhi = *(const v8h*)(bp + 16);
        v16h bf;
        #pragma unroll
        for (int e = 0; e < 8; ++e) { bf[e] = blo[e]; bf[8 + e] = bhi[e]; }
        acc = __builtin_amdgcn_wmma_f32_16x16x32_f16(
            false, af, false, bf, (short)0, acc, false, false);
    }
    if (r < 10) {
        float bias = predb[r];
        #pragma unroll
        for (int i = 0; i < 8; ++i) {
            int mm = mbase + i + ((lane >> 4) << 3);
            out[mm * 10 + r] = acc[i] + bias;
        }
    }
}

// ================= launcher =================
extern "C" void kernel_launch(void* const* d_in, const int* in_sizes, int n_in,
                              void* d_out, int out_size, void* d_ws, size_t ws_size,
                              hipStream_t stream) {
    const float* x      = (const float*)d_in[0];
    const float* w1     = (const float*)d_in[1];
    const float* b1     = (const float*)d_in[2];
    const float* w2     = (const float*)d_in[3];
    const float* c2b    = (const float*)d_in[4];
    const float* pred_w = (const float*)d_in[5];
    const float* pred_b = (const float*)d_in[6];
    float* out = (float*)d_out;

    char* ws = (char*)d_ws;
    float*     theta = (float*)(ws + OFF_THETA);
    _Float16*  h1    = (_Float16*)(ws + OFF_H1);
    _Float16*  out2  = (_Float16*)(ws + OFF_OUT2);
    _Float16*  w2t   = (_Float16*)(ws + OFF_W2T);
    _Float16*  pwt   = (_Float16*)(ws + OFF_PWT);

    k_prep_w2t<<<(C2 * K2 + 255) / 256, 256, 0, stream>>>(w2, w2t);
    k_prep_pwt<<<(NPAD * KF + 255) / 256, 256, 0, stream>>>(pred_w, pwt);
    k_angles<<<B_IMG, 256, 0, stream>>>(x, theta);
    k_rot_conv1<<<B_IMG, 256, 0, stream>>>(x, theta, w1, b1, h1);
    k_conv2_wmma<<<M2 / 16 / 8, 256, 0, stream>>>(h1, w2t, c2b, out2);   // 3136 blocks
    k_final_wmma<<<(B_IMG / 16) / 8, 256, 0, stream>>>(out2, pwt, pred_b, out); // 16 blocks
}
